// GCNNet_13099650253563
// MI455X (gfx1250) — matrix operand order repaired and checked by reading, not compiled, not run on hardware
//
#include <hip/hip_runtime.h>
#include <cstdint>
#include <cstddef>

#define NN 50000
#define NE 1600000

typedef __bf16   bf16_t;
typedef bf16_t   v16bf __attribute__((ext_vector_type(16)));
typedef float    v8f   __attribute__((ext_vector_type(8)));
typedef uint32_t v8u   __attribute__((ext_vector_type(8)));

// ---------------- small helpers ----------------
__device__ __forceinline__ uint16_t f32_to_bf16(float f) {
  uint32_t u = __float_as_uint(f);
  uint32_t r = u + 0x7FFFu + ((u >> 16) & 1u);   // round-to-nearest-even
  return (uint16_t)(r >> 16);
}

// order-preserving float->uint encoding for atomic max
__device__ __forceinline__ float decode_agg(uint32_t e) {
  if (e == 0u) return 0.0f;                          // sentinel: no incoming edge
  uint32_t u = (e & 0x80000000u) ? (e & 0x7FFFFFFFu) : ~e;
  if ((u & 0x7F800000u) == 0x7F800000u) return 0.0f; // non-finite -> 0 (reference)
  return __uint_as_float(u);
}

// ---------------- elementwise kernels ----------------
__global__ void cvt_bf16_k(const float* __restrict__ in, uint16_t* __restrict__ out, int n) {
  int i = blockIdx.x * blockDim.x + threadIdx.x;
  if (i < n) out[i] = f32_to_bf16(in[i]);
}

__global__ void fill_u32_k(uint32_t* __restrict__ p, uint32_t v, long long n) {
  long long i = (long long)blockIdx.x * blockDim.x + threadIdx.x;
  if (i < n) p[i] = v;
}

// Pre-swizzle f32 weights [K x Ncols] into WMMA bf16 B-fragment order.
// out[((kt*ntilesN + nt)*32 + lane)*8 + v] packs bf16 pair {B[k][n], B[k+1][n]}
// with n = nt*16 + (lane&15), k = kt*32 + (lane>>4)*16 + 2*v. Columns >= Ncols -> 0.
__global__ void swizzle_w_k(const float* __restrict__ W, uint32_t* __restrict__ out,
                            int K, int Ncols, int ntilesN) {
  int gid = blockIdx.x * blockDim.x + threadIdx.x;
  int total = (K >> 5) * ntilesN * 256;
  if (gid >= total) return;
  int v    = gid & 7;
  int lane = (gid >> 3) & 31;
  int tile = gid >> 8;
  int nt   = tile % ntilesN;
  int kt   = tile / ntilesN;
  int half = lane >> 4;
  int n    = nt * 16 + (lane & 15);
  int k    = (kt << 5) + half * 16 + v * 2;
  uint32_t lo = 0, hi = 0;
  if (n < Ncols) {
    lo = f32_to_bf16(W[(size_t)k * Ncols + n]);
    hi = f32_to_bf16(W[(size_t)(k + 1) * Ncols + n]);
  }
  out[gid] = lo | (hi << 16);
}

// ---------------- WMMA GEMM: C[M x Ncols] = A(bf16,row-major,ld=K) @ Bswz ----------------
// One wave per 16x(16*NT) output slab; K-loop in steps of 32 with v_wmma_f32_16x16x32_bf16.
template <int NT>
__global__ void __launch_bounds__(256)
gemm_wmma_k(const uint16_t* __restrict__ A, const uint32_t* __restrict__ Bsw,
            float* __restrict__ C, int K, int Ncols, int ntilesN, int tilesM) {
  const int wave = (int)((blockIdx.x * blockDim.x + threadIdx.x) >> 5);
  const int lane = threadIdx.x & 31;
  const int ngrp = ntilesN / NT;
  const int tm   = wave / ngrp;
  const int tg   = wave - tm * ngrp;
  if (tm >= tilesM) return;            // uniform per wave

  const int half = lane >> 4;
  const int mrow = tm * 16 + (lane & 15);
  const uint16_t* Arow = A + (size_t)mrow * K;

  v8f acc[NT];
#pragma unroll
  for (int t = 0; t < NT; ++t)
#pragma unroll
    for (int j = 0; j < 8; ++j) acc[t][j] = 0.0f;

  const int ktiles = K >> 5;
  for (int kt = 0; kt < ktiles; ++kt) {
    const int k0 = kt << 5;
    v8u au;
#pragma unroll
    for (int v = 0; v < 8; ++v) {
      // A 16x32 bf16 fragment layout (matches Bswz): lanes 0-15 K=0-15; lanes 16-31 K=16-31
      const int kk = k0 + (v << 1) + (half << 4);
      au[v] = *(const uint32_t*)(Arow + kk);
    }
    v16bf a = __builtin_bit_cast(v16bf, au);
#pragma unroll
    for (int t = 0; t < NT; ++t) {
      const uint32_t* bp = Bsw + ((((size_t)kt * ntilesN) + (tg * NT + t)) * 32 + lane) * 8;
      v8u bu;
#pragma unroll
      for (int v = 0; v < 8; ++v) bu[v] = bp[v];
      v16bf b = __builtin_bit_cast(v16bf, bu);
      acc[t] = __builtin_amdgcn_wmma_f32_16x16x32_bf16(
          false, a, false, b, (short)0, acc[t], false, false);
    }
  }

  // C/D layout: lane -> n = lane&15; vgpr r -> m = tm*16 + (lane>>4)*8 + r
  const int m0 = tm * 16 + (half << 3);
#pragma unroll
  for (int t = 0; t < NT; ++t) {
    const int n = (tg * NT + t) * 16 + (lane & 15);
    if (n < Ncols) {
#pragma unroll
      for (int r = 0; r < 8; ++r)
        C[(size_t)(m0 + r) * Ncols + n] = acc[t][r];
    }
  }
}

// ---------------- edge scatter-max (L2-resident atomics) ----------------
template <int DOUT>
__global__ void edge_scatter_k(const int* __restrict__ src, const int* __restrict__ dst,
                               const float* __restrict__ m, uint32_t* __restrict__ agg,
                               long long total) {
  long long gid = (long long)blockIdx.x * blockDim.x + threadIdx.x;
  if (gid >= total) return;
  const int e = (int)(gid / DOUT);
  const int c = (int)(gid - (long long)e * DOUT);
  const float val = m[(size_t)src[e] * DOUT + c];
  const uint32_t u = __float_as_uint(val);
  const uint32_t enc = (u & 0x80000000u) ? ~u : (u | 0x80000000u);
  atomicMax(agg + (size_t)dst[e] * DOUT + c, enc);
}

// ---------------- combine variants (dout == 128 unless noted) ----------------
// intermediate layers: out = max_agg + b + root; accumulate per-channel sum/sumsq
__global__ void combine_stats_k(const uint32_t* __restrict__ agg, const float* __restrict__ mroot,
                                const float* __restrict__ bias, float* __restrict__ out,
                                float* __restrict__ ssum, float* __restrict__ ssq, int N) {
  const int tid = threadIdx.x;
  const int c = tid & 127;
  float s = 0.f, s2 = 0.f;
  for (int i = blockIdx.x * 2 + (tid >> 7); i < N; i += gridDim.x * 2) {
    size_t idx = (size_t)i * 128 + c;
    float v = decode_agg(agg[idx]) + bias[c] + mroot[idx];
    out[idx] = v;
    s += v; s2 += v * v;
  }
  __shared__ float sh[512];
  sh[tid] = s; sh[256 + tid] = s2;
  __syncthreads();
  if (tid < 128) {
    atomicAdd(&ssum[c], sh[tid] + sh[tid + 128]);
    atomicAdd(&ssq[c],  sh[256 + tid] + sh[256 + tid + 128]);
  }
}

// global-head final layer: only need column sums (for mean over nodes)
__global__ void combine_gsum_k(const uint32_t* __restrict__ agg, const float* __restrict__ mroot,
                               const float* __restrict__ bias, float* __restrict__ ssum, int N) {
  const int tid = threadIdx.x;
  const int c = tid & 127;
  float s = 0.f;
  for (int i = blockIdx.x * 2 + (tid >> 7); i < N; i += gridDim.x * 2) {
    size_t idx = (size_t)i * 128 + c;
    s += decode_agg(agg[idx]) + bias[c] + mroot[idx];
  }
  __shared__ float sh[256];
  sh[tid] = s;
  __syncthreads();
  if (tid < 128) atomicAdd(&ssum[c], sh[tid] + sh[tid + 128]);
}

// head/skip final layers: write bf16 directly into concat buffer column block
__global__ void combine_bf16cols_k(const uint32_t* __restrict__ agg, const float* __restrict__ mroot,
                                   const float* __restrict__ bias, uint16_t* __restrict__ xcat,
                                   int coloff, int N) {
  int gid = blockIdx.x * blockDim.x + threadIdx.x;
  if (gid >= N * 128) return;
  int c = gid & 127, i = gid >> 7;
  float v = decode_agg(agg[gid]) + bias[c] + mroot[gid];
  xcat[(size_t)i * 384 + coloff + c] = f32_to_bf16(v);
}

// tail final layer (dout = 3): tanh(v)*0.5 straight to d_out
__global__ void combine_final_k(const uint32_t* __restrict__ agg, const float* __restrict__ mroot,
                                const float* __restrict__ bias, float* __restrict__ out, int N) {
  int gid = blockIdx.x * blockDim.x + threadIdx.x;
  if (gid >= N * 3) return;
  int c = gid % 3;
  float v = decode_agg(agg[gid]) + bias[c] + mroot[gid];
  out[gid] = tanhf(v) * 0.5f;
}

// InstanceNorm (no affine) + LeakyReLU(0.02), emit bf16 for next layer's GEMM A
__global__ void norm_act_k(const float* __restrict__ in, const float* __restrict__ ssum,
                           const float* __restrict__ ssq, uint16_t* __restrict__ xa, int N) {
  int gid = blockIdx.x * blockDim.x + threadIdx.x;
  if (gid >= N * 128) return;
  int c = gid & 127;
  const float invN = 1.0f / (float)N;
  float mu  = ssum[c] * invN;
  float var = ssq[c] * invN - mu * mu;
  float r   = rsqrtf(var + 1e-5f);
  float v   = (in[gid] - mu) * r;
  v = v >= 0.0f ? v : 0.02f * v;
  xa[gid] = f32_to_bf16(v);
}

// broadcast mean(global head) into xcat columns 256..383
__global__ void broadcast_k(const float* __restrict__ ssum, uint16_t* __restrict__ xcat, int N) {
  int gid = blockIdx.x * blockDim.x + threadIdx.x;
  if (gid >= N * 128) return;
  int c = gid & 127, i = gid >> 7;
  xcat[(size_t)i * 384 + 256 + c] = f32_to_bf16(ssum[c] * (1.0f / (float)N));
}

// ---------------- host orchestration ----------------
static void launch_layer(hipStream_t stream, const uint16_t* Abf, int din, int dout,
                         const float* Wrel, const float* Wroot,
                         uint32_t* wswzR, uint32_t* wswzS,
                         float* mrel, float* mroot, uint32_t* agg,
                         const int* esrc, const int* edst) {
  const int ktiles = din >> 5;
  const int ntN = (dout + 15) >> 4;
  const int swzTot = ktiles * ntN * 256;
  swizzle_w_k<<<(swzTot + 255) / 256, 256, 0, stream>>>(Wrel,  wswzR, din, dout, ntN);
  swizzle_w_k<<<(swzTot + 255) / 256, 256, 0, stream>>>(Wroot, wswzS, din, dout, ntN);

  const int tilesM = NN / 16;  // 3125, exact
  if ((ntN & 3) == 0) {
    const int waves = tilesM * (ntN >> 2);
    dim3 grid((unsigned)((waves * 32 + 255) / 256));
    gemm_wmma_k<4><<<grid, 256, 0, stream>>>(Abf, wswzR, mrel,  din, dout, ntN, tilesM);
    gemm_wmma_k<4><<<grid, 256, 0, stream>>>(Abf, wswzS, mroot, din, dout, ntN, tilesM);
  } else {
    const int waves = tilesM * ntN;
    dim3 grid((unsigned)((waves * 32 + 255) / 256));
    gemm_wmma_k<1><<<grid, 256, 0, stream>>>(Abf, wswzR, mrel,  din, dout, ntN, tilesM);
    gemm_wmma_k<1><<<grid, 256, 0, stream>>>(Abf, wswzS, mroot, din, dout, ntN, tilesM);
  }

  const long long aggN = (long long)NN * dout;
  fill_u32_k<<<(int)((aggN + 255) / 256), 256, 0, stream>>>(agg, 0u, aggN);
  const long long tot = (long long)NE * dout;
  if (dout == 128)
    edge_scatter_k<128><<<(int)((tot + 255) / 256), 256, 0, stream>>>(esrc, edst, mrel, agg, tot);
  else
    edge_scatter_k<3><<<(int)((tot + 255) / 256), 256, 0, stream>>>(esrc, edst, mrel, agg, tot);
}

static void intermediate_post(hipStream_t stream, const uint32_t* agg, const float* mroot,
                              const float* bias, float* outbuf, float* ssum, float* ssq,
                              uint16_t* xa) {
  fill_u32_k<<<1, 256, 0, stream>>>((uint32_t*)ssum, 0u, 256);
  combine_stats_k<<<1024, 256, 0, stream>>>(agg, mroot, bias, outbuf, ssum, ssq, NN);
  norm_act_k<<<(NN * 128 + 255) / 256, 256, 0, stream>>>(outbuf, ssum, ssq, xa, NN);
}

extern "C" void kernel_launch(void* const* d_in, const int* in_sizes, int n_in,
                              void* d_out, int out_size, void* d_ws, size_t ws_size,
                              hipStream_t stream) {
  (void)in_sizes; (void)n_in; (void)out_size; (void)ws_size;
  const float* x  = (const float*)d_in[0];
  const int*   ei = (const int*)d_in[1];
  const int* esrc = ei;
  const int* edst = ei + NE;

  // workspace carve (256B aligned)
  char* p = (char*)d_ws;
  auto carve = [&](size_t b) -> void* { void* r = (void*)p; p += (b + 255) & ~(size_t)255; return r; };
  uint16_t* x0    = (uint16_t*)carve((size_t)NN * 64 * 2);    // x in bf16
  uint16_t* xa    = (uint16_t*)carve((size_t)NN * 128 * 2);   // intermediate activations (bf16)
  uint16_t* xcat  = (uint16_t*)carve((size_t)NN * 384 * 2);   // concat buffer (bf16)
  uint32_t* wswzR = (uint32_t*)carve((size_t)12 * 8 * 32 * 8 * 4);
  uint32_t* wswzS = (uint32_t*)carve((size_t)12 * 8 * 32 * 8 * 4);
  float*    mrel  = (float*)carve((size_t)NN * 128 * 4);      // x@Wr (messages)
  float*    mroot = (float*)carve((size_t)NN * 128 * 4);      // x@Ws
  uint32_t* agg   = (uint32_t*)carve((size_t)NN * 128 * 4);   // encoded scatter-max
  float*    stats = (float*)carve(1024);
  float* ssum = stats;
  float* ssq  = stats + 128;
  float* outbuf = mrel;  // mrel is dead after edge scatter -> reuse as fp32 layer output

  cvt_bf16_k<<<(NN * 64 + 255) / 256, 256, 0, stream>>>(x, x0, NN * 64);

  // ---- head: 5 layers, params d_in[2..16] ----
  const uint16_t* A = x0; int din = 64;
  for (int l = 0; l < 5; ++l) {
    const float* Wr = (const float*)d_in[2 + 3 * l];
    const float* Ws = (const float*)d_in[3 + 3 * l];
    const float* b  = (const float*)d_in[4 + 3 * l];
    launch_layer(stream, A, din, 128, Wr, Ws, wswzR, wswzS, mrel, mroot, agg, esrc, edst);
    if (l < 4) {
      intermediate_post(stream, agg, mroot, b, outbuf, ssum, ssq, xa);
      A = xa; din = 128;
    } else {
      combine_bf16cols_k<<<(NN * 128 + 255) / 256, 256, 0, stream>>>(agg, mroot, b, xcat, 0, NN);
    }
  }

  // ---- skip: 1 layer, params d_in[17..19] ----
  launch_layer(stream, x0, 64, 128, (const float*)d_in[17], (const float*)d_in[18],
               wswzR, wswzS, mrel, mroot, agg, esrc, edst);
  combine_bf16cols_k<<<(NN * 128 + 255) / 256, 256, 0, stream>>>(
      agg, mroot, (const float*)d_in[19], xcat, 128, NN);

  // ---- global head: 2 layers, params d_in[20..25]; mean -> broadcast ----
  launch_layer(stream, x0, 64, 128, (const float*)d_in[20], (const float*)d_in[21],
               wswzR, wswzS, mrel, mroot, agg, esrc, edst);
  intermediate_post(stream, agg, mroot, (const float*)d_in[22], outbuf, ssum, ssq, xa);
  launch_layer(stream, xa, 128, 128, (const float*)d_in[23], (const float*)d_in[24],
               wswzR, wswzS, mrel, mroot, agg, esrc, edst);
  fill_u32_k<<<1, 256, 0, stream>>>((uint32_t*)ssum, 0u, 256);
  combine_gsum_k<<<1024, 256, 0, stream>>>(agg, mroot, (const float*)d_in[25], ssum, NN);
  broadcast_k<<<(NN * 128 + 255) / 256, 256, 0, stream>>>(ssum, xcat, NN);

  // ---- tail: 2 layers, params d_in[26..31]; input = xcat [N,384] ----
  launch_layer(stream, xcat, 384, 128, (const float*)d_in[26], (const float*)d_in[27],
               wswzR, wswzS, mrel, mroot, agg, esrc, edst);
  intermediate_post(stream, agg, mroot, (const float*)d_in[28], outbuf, ssum, ssq, xa);
  launch_layer(stream, xa, 128, 3, (const float*)d_in[29], (const float*)d_in[30],
               wswzR, wswzS, mrel, mroot, agg, esrc, edst);
  combine_final_k<<<(NN * 3 + 255) / 256, 256, 0, stream>>>(
      agg, mroot, (const float*)d_in[31], (float*)d_out, NN);
}
